// DualStreamBlock_9878424781299
// MI455X (gfx1250) — compile-verified, matrix operand order checked by reading, not run on hardware
//
#include <hip/hip_runtime.h>
#include <hip/hip_bf16.h>

typedef __attribute__((ext_vector_type(16))) _Float16 v16h;
typedef __attribute__((ext_vector_type(8)))  float    v8f;
typedef __attribute__((ext_vector_type(4)))  int      v4i;

#define CH   256
#define WSZ  256
#define NH   8
#define HD   32
#define HID  512
#define SCALE 0.17677669529663687f  // 32^-0.5

#define KKOF(e, hh) ((((e) >> 3) << 4) + (hh) * 8 + ((e) & 7))

__device__ __forceinline__ v8f v8f_zero() {
  v8f z;
#pragma unroll
  for (int i = 0; i < 8; ++i) z[i] = 0.f;
  return z;
}

__device__ __forceinline__ v8f wmma32(v16h a, v16h b, v8f c) {
  return __builtin_amdgcn_wmma_f32_16x16x32_f16(false, a, false, b, (short)0, c,
                                                false, false);
}

// fragment loader: element e lives at p + KKOF(e,hh)*kstride (f16 source)
__device__ __forceinline__ v16h frag_f16(const _Float16* __restrict__ p,
                                         size_t kstride, int hh) {
  v16h f;
#pragma unroll
  for (int e = 0; e < 16; ++e) f[e] = p[(size_t)KKOF(e, hh) * kstride];
  return f;
}

// fragment from pre-packed B: tiles of [32 lanes][16 f16], 32B-aligned vector load
__device__ __forceinline__ v16h frag_pk(const _Float16* __restrict__ P,
                                        int cols16, int kc, int ctg, int lane) {
  size_t tile = (size_t)(kc >> 5) * cols16 + ctg;
  return *(const v16h*)(P + (tile << 9) + ((size_t)lane << 4));
}

// ------------- pack f32 weight [K][cols] -> WMMA B-fragment tiles (f16) ------
__global__ __launch_bounds__(256) void pack_b_kernel(
    const float* __restrict__ B, _Float16* __restrict__ P, int K, int cols) {
  int tile = blockIdx.x;                 // (K/32) * (cols/16) tiles
  int cols16 = cols >> 4;
  int kc = (tile / cols16) << 5;
  int ct = tile % cols16;
  for (int i = threadIdx.x; i < 512; i += 256) {
    int lane = i >> 4, e = i & 15;
    int n = lane & 15, hh = lane >> 4;
    P[((size_t)tile << 9) + i] =
        (_Float16)B[(size_t)(kc + KKOF(e, hh)) * cols + ct * 16 + n];
  }
}

// ---------------- LayerNorm (f32 in -> f16 out) ----------------
__global__ __launch_bounds__(256) void ln_f32_kernel(
    const float* __restrict__ x, const float* __restrict__ g,
    const float* __restrict__ b, _Float16* __restrict__ out, int cols, int relu) {
  int row = blockIdx.x, tid = threadIdx.x;
  const float* xr = x + (size_t)row * cols;
  __shared__ float red[8];
  float s = 0.f;
  for (int c = tid; c < cols; c += 256) s += xr[c];
  for (int off = 16; off > 0; off >>= 1) s += __shfl_xor(s, off, 32);
  if ((tid & 31) == 0) red[tid >> 5] = s;
  __syncthreads();
  float tot = 0.f;
  for (int i = 0; i < 8; ++i) tot += red[i];
  float mean = tot / cols;
  __syncthreads();
  float vs = 0.f;
  for (int c = tid; c < cols; c += 256) { float d = xr[c] - mean; vs += d * d; }
  for (int off = 16; off > 0; off >>= 1) vs += __shfl_xor(vs, off, 32);
  if ((tid & 31) == 0) red[tid >> 5] = vs;
  __syncthreads();
  float vt = 0.f;
  for (int i = 0; i < 8; ++i) vt += red[i];
  float rstd = rsqrtf(vt / cols + 1e-5f);
  for (int c = tid; c < cols; c += 256) {
    float val = (xr[c] - mean) * rstd * g[c] + b[c];
    if (relu) val = fmaxf(val, 0.f);
    out[(size_t)row * cols + c] = (_Float16)val;
  }
}

// ---------------- LayerNorm (f16 in -> f16 out) ----------------
__global__ __launch_bounds__(256) void ln_f16_kernel(
    const _Float16* __restrict__ x, const float* __restrict__ g,
    const float* __restrict__ b, _Float16* __restrict__ out, int cols, int relu) {
  int row = blockIdx.x, tid = threadIdx.x;
  const _Float16* xr = x + (size_t)row * cols;
  __shared__ float red[8];
  float s = 0.f;
  for (int c = tid; c < cols; c += 256) s += (float)xr[c];
  for (int off = 16; off > 0; off >>= 1) s += __shfl_xor(s, off, 32);
  if ((tid & 31) == 0) red[tid >> 5] = s;
  __syncthreads();
  float tot = 0.f;
  for (int i = 0; i < 8; ++i) tot += red[i];
  float mean = tot / cols;
  __syncthreads();
  float vs = 0.f;
  for (int c = tid; c < cols; c += 256) { float d = (float)xr[c] - mean; vs += d * d; }
  for (int off = 16; off > 0; off >>= 1) vs += __shfl_xor(vs, off, 32);
  if ((tid & 31) == 0) red[tid >> 5] = vs;
  __syncthreads();
  float vt = 0.f;
  for (int i = 0; i < 8; ++i) vt += red[i];
  float rstd = rsqrtf(vt / cols + 1e-5f);
  for (int c = tid; c < cols; c += 256) {
    float val = ((float)xr[c] - mean) * rstd * g[c] + b[c];
    if (relu) val = fmaxf(val, 0.f);
    out[(size_t)row * cols + c] = (_Float16)val;
  }
}

// ---------------- fused window attention: one block per (window, head) -------
__global__ __launch_bounds__(256) void win_attn_kernel(
    const _Float16* __restrict__ xh, const int* __restrict__ win_idx,
    const unsigned char* __restrict__ win_mask,
    const _Float16* __restrict__ qkvw_p,  // packed [256][768] weight
    _Float16* __restrict__ attn_out, int M) {
  __shared__ _Float16 qs[WSZ * HD];   // 16KB
  __shared__ _Float16 ks[WSZ * HD];   // 16KB
  __shared__ _Float16 vs[WSZ * HD];   // 16KB
  __shared__ _Float16 un[WSZ * HD];   // 16KB union: xtile (A) / P staging (B)

  int m = blockIdx.x / NH;
  int h = blockIdx.x % NH;
  int tid = threadIdx.x;
  int wave = tid >> 5, lane = tid & 31;
  int r = lane & 15, hh = lane >> 4;
  const int* idx = win_idx + (size_t)m * WSZ;

  // ---- Stage A: q,k,v = gather(xh) @ qkv_w slices ----
  // wave owns ct = wave&1 and row-tiles {wave>>1, +4, +8, +12} for q,k,v.
  int ctw = wave & 1;
  int rtb = wave >> 1;
  v8f acc[12];                           // [mat*4 + rr]
#pragma unroll
  for (int i = 0; i < 12; ++i) acc[i] = v8f_zero();

  for (int kc = 0; kc < CH; kc += 32) {
    { // cooperative gather of x-tile [256 rows x 32 cols] f16 (16B vector copies)
      int row = tid;
      int gi = idx[row];
      const v4i* src = (const v4i*)(xh + (size_t)gi * CH + kc);
      v4i* dst = (v4i*)(un + row * 32);
#pragma unroll
      for (int j = 0; j < 4; ++j) dst[j] = src[j];
    }
    __syncthreads();
    v16h afr[4];
#pragma unroll
    for (int rr = 0; rr < 4; ++rr) {
      int rt = rtb + rr * 4;
      afr[rr] = frag_f16(un + (rt * 16 + r) * 32, 1, hh);
    }
#pragma unroll
    for (int mat = 0; mat < 3; ++mat) {
      v16h b = frag_pk(qkvw_p, 48, kc, mat * 16 + h * 2 + ctw, lane);
#pragma unroll
      for (int rr = 0; rr < 4; ++rr)
        acc[mat * 4 + rr] = wmma32(afr[rr], b, acc[mat * 4 + rr]);
    }
    __syncthreads();
  }
  // store q (pre-scaled), k, v to LDS
#pragma unroll
  for (int mat = 0; mat < 3; ++mat) {
    _Float16* dst = (mat == 0) ? qs : (mat == 1) ? ks : vs;
    float sc = (mat == 0) ? SCALE : 1.0f;
#pragma unroll
    for (int rr = 0; rr < 4; ++rr) {
      int rt = rtb + rr * 4;
#pragma unroll
      for (int v = 0; v < 8; ++v) {
        int row = rt * 16 + v + 8 * hh;
        dst[row * HD + ctw * 16 + r] = (_Float16)(acc[mat * 4 + rr][v] * sc);
      }
    }
  }
  __syncthreads();

  // ---- Stage B: flash attention; wave owns row-groups {wave, wave+8} ----
  const unsigned char* msk = win_mask + (size_t)m * WSZ;
  _Float16* st = un + wave * (16 * 32);  // per-wave P staging (f16)

  for (int rg = wave; rg < 16; rg += 8) {
    int row_base = rg * 16;
    v8f o0 = v8f_zero(), o1 = v8f_zero();
    float mrow[8], lrow[8];
#pragma unroll
    for (int v = 0; v < 8; ++v) { mrow[v] = -1e30f; lrow[v] = 0.f; }
    v16h aq = frag_f16(qs + (row_base + r) * HD, 1, hh);

    for (int cc = 0; cc < WSZ; cc += 32) {
      v16h b0 = frag_f16(ks + (cc + r) * HD, 1, hh);
      v16h b1 = frag_f16(ks + (cc + 16 + r) * HD, 1, hh);
      v8f s0 = wmma32(aq, b0, v8f_zero());
      v8f s1 = wmma32(aq, b1, v8f_zero());
      bool m0 = msk[cc + r] != 0;
      bool m1 = msk[cc + 16 + r] != 0;
#pragma unroll
      for (int v = 0; v < 8; ++v) {
        s0[v] = m0 ? s0[v] : -1e30f;
        s1[v] = m1 ? s1[v] : -1e30f;
      }
#pragma unroll
      for (int v = 0; v < 8; ++v) {
        float t = fmaxf(s0[v], s1[v]);
        for (int off = 1; off < 16; off <<= 1) t = fmaxf(t, __shfl_xor(t, off, 16));
        float nm = fmaxf(mrow[v], t);
        float sf = __expf(mrow[v] - nm);
        mrow[v] = nm;
        float p0 = __expf(s0[v] - nm);
        float p1 = __expf(s1[v] - nm);
        s0[v] = p0; s1[v] = p1;
        float rs = p0 + p1;
        for (int off = 1; off < 16; off <<= 1) rs += __shfl_xor(rs, off, 16);
        lrow[v] = lrow[v] * sf + rs;
        o0[v] *= sf;
        o1[v] *= sf;
      }
      // stage P (C-layout -> row-major f16) through per-wave LDS
#pragma unroll
      for (int v = 0; v < 8; ++v) {
        int rowp = v + 8 * hh;
        st[rowp * 32 + r] = (_Float16)s0[v];
        st[rowp * 32 + 16 + r] = (_Float16)s1[v];
      }
      asm volatile("s_wait_dscnt 0x0" ::: "memory");
      v16h ap = frag_f16(st + r * 32, 1, hh);
      v16h bv0 = frag_f16(vs + cc * HD + r, HD, hh);
      v16h bv1 = frag_f16(vs + cc * HD + 16 + r, HD, hh);
      o0 = wmma32(ap, bv0, o0);
      o1 = wmma32(ap, bv1, o1);
    }
#pragma unroll
    for (int v = 0; v < 8; ++v) {
      float inv = 1.0f / fmaxf(lrow[v], 1e-20f);
      int row = row_base + v + 8 * hh;
      size_t base = ((size_t)m * WSZ + row) * CH + h * HD;
      attn_out[base + r] = (_Float16)(o0[v] * inv);
      attn_out[base + 16 + r] = (_Float16)(o1[v] * inv);
    }
  }
}

// ---------------- centroid compute ----------------
__global__ __launch_bounds__(256) void cent_kernel(
    const _Float16* __restrict__ xh, const int* __restrict__ win_idx,
    const unsigned char* __restrict__ win_mask, float* __restrict__ cent, int M) {
  int m = blockIdx.x, c = threadIdx.x;
  float s = 0.f, cnt = 0.f;
  for (int w = 0; w < WSZ; ++w) {
    if (win_mask[(size_t)m * WSZ + w]) {
      s += (float)xh[(size_t)win_idx[(size_t)m * WSZ + w] * CH + c];
      cnt += 1.f;
    }
  }
  cent[(size_t)m * CH + c] = s / fmaxf(cnt, 1.f);
}

// ---------------- generic WMMA GEMM: Out(f16) = A(f16)@Bp + bias -------------
__global__ __launch_bounds__(256) void gemm_f16out_kernel(
    const _Float16* __restrict__ A, const _Float16* __restrict__ Bp,
    const float* __restrict__ bias, _Float16* __restrict__ Out, int rows, int K,
    int cols) {
  int rb = blockIdx.x * 64, cb = blockIdx.y * 64;
  int tid = threadIdx.x, wave = tid >> 5, lane = tid & 31;
  int r = lane & 15, hh = lane >> 4;
  int rt = wave >> 1;
  int ct0 = (wave & 1) * 2;
  int cols16 = cols >> 4;
  v8f a0 = v8f_zero(), a1 = v8f_zero();
  int arow = rb + rt * 16 + r;
  if (arow >= rows) arow = rows - 1;
  for (int kc = 0; kc < K; kc += 32) {
    v16h af = frag_f16(A + (size_t)arow * K + kc, 1, hh);
    v16h b0 = frag_pk(Bp, cols16, kc, (cb >> 4) + ct0, lane);
    v16h b1 = frag_pk(Bp, cols16, kc, (cb >> 4) + ct0 + 1, lane);
    a0 = wmma32(af, b0, a0);
    a1 = wmma32(af, b1, a1);
  }
#pragma unroll
  for (int j = 0; j < 2; ++j) {
    v8f acc = j ? a1 : a0;
    int col = cb + (ct0 + j) * 16 + r;
    float bv = bias ? bias[col] : 0.f;
#pragma unroll
    for (int v = 0; v < 8; ++v) {
      int row = rb + rt * 16 + v + 8 * hh;
      if (row < rows) Out[(size_t)row * cols + col] = (_Float16)(acc[v] + bv);
    }
  }
}

// ---- Out(f32) = res(f32) + A(f16)@Bp + bias ----
__global__ __launch_bounds__(256) void gemm_f32res_kernel(
    const _Float16* __restrict__ A, const _Float16* __restrict__ Bp,
    const float* __restrict__ bias, const float* __restrict__ res,
    float* __restrict__ Out, int rows, int K, int cols) {
  int rb = blockIdx.x * 64, cb = blockIdx.y * 64;
  int tid = threadIdx.x, wave = tid >> 5, lane = tid & 31;
  int r = lane & 15, hh = lane >> 4;
  int rt = wave >> 1;
  int ct0 = (wave & 1) * 2;
  int cols16 = cols >> 4;
  v8f a0 = v8f_zero(), a1 = v8f_zero();
  int arow = rb + rt * 16 + r;
  if (arow >= rows) arow = rows - 1;
  for (int kc = 0; kc < K; kc += 32) {
    v16h af = frag_f16(A + (size_t)arow * K + kc, 1, hh);
    v16h b0 = frag_pk(Bp, cols16, kc, (cb >> 4) + ct0, lane);
    v16h b1 = frag_pk(Bp, cols16, kc, (cb >> 4) + ct0 + 1, lane);
    a0 = wmma32(af, b0, a0);
    a1 = wmma32(af, b1, a1);
  }
#pragma unroll
  for (int j = 0; j < 2; ++j) {
    v8f acc = j ? a1 : a0;
    int col = cb + (ct0 + j) * 16 + r;
    float bv = bias ? bias[col] : 0.f;
#pragma unroll
    for (int v = 0; v < 8; ++v) {
      int row = rb + rt * 16 + v + 8 * hh;
      if (row < rows) {
        size_t o = (size_t)row * cols + col;
        Out[o] = res[o] + acc[v] + bv;
      }
    }
  }
}

// ---------------- centroid attention (tiny; VALU online softmax) -------------
__global__ __launch_bounds__(256) void cattn_kernel(
    const _Float16* __restrict__ cqkv, _Float16* __restrict__ co, int M) {
  int h = blockIdx.x;
  int tid = threadIdx.x;
  for (int q = tid; q < M; q += 256) {
    float qv[HD];
#pragma unroll
    for (int d = 0; d < HD; ++d)
      qv[d] = (float)cqkv[(size_t)q * (3 * CH) + h * HD + d] * SCALE;
    float mx = -1e30f, l = 0.f;
    float accv[HD];
#pragma unroll
    for (int d = 0; d < HD; ++d) accv[d] = 0.f;
    for (int kk = 0; kk < M; ++kk) {
      const _Float16* kp = cqkv + (size_t)kk * (3 * CH) + CH + h * HD;
      const _Float16* vp = cqkv + (size_t)kk * (3 * CH) + 2 * CH + h * HD;
      float s = 0.f;
#pragma unroll
      for (int d = 0; d < HD; ++d) s += qv[d] * (float)kp[d];
      float nm = fmaxf(mx, s);
      float sf = __expf(mx - nm);
      float p = __expf(s - nm);
      mx = nm;
      l = l * sf + p;
#pragma unroll
      for (int d = 0; d < HD; ++d) accv[d] = accv[d] * sf + p * (float)vp[d];
    }
#pragma unroll
    for (int d = 0; d < HD; ++d)
      co[(size_t)q * CH + h * HD + d] = (_Float16)(accv[d] / l);
  }
}

// ------- window proj + global-ctx broadcast + overlap scatter (atomics) ------
__global__ __launch_bounds__(256) void win_proj_scatter_kernel(
    const _Float16* __restrict__ attn_out, const _Float16* __restrict__ projw_p,
    const float* __restrict__ proj_b, const float* __restrict__ cent_out,
    const int* __restrict__ win_idx, const unsigned char* __restrict__ win_mask,
    float* __restrict__ y, int M) {
  int m = blockIdx.x, cq = blockIdx.y;
  int tid = threadIdx.x, wave = tid >> 5, lane = tid & 31;
  int r = lane & 15, hh = lane >> 4;
  v8f acc[8];
#pragma unroll
  for (int j = 0; j < 8; ++j) acc[j] = v8f_zero();
  for (int kc = 0; kc < CH; kc += 32) {
    v16h a0 = frag_f16(attn_out + ((size_t)m * WSZ + (wave * 2) * 16 + r) * CH + kc, 1, hh);
    v16h a1 = frag_f16(attn_out + ((size_t)m * WSZ + (wave * 2 + 1) * 16 + r) * CH + kc, 1, hh);
    v16h bfr[4];
#pragma unroll
    for (int ct = 0; ct < 4; ++ct)
      bfr[ct] = frag_pk(projw_p, 16, kc, cq * 4 + ct, lane);
#pragma unroll
    for (int j = 0; j < 8; ++j)
      acc[j] = wmma32((j >> 2) ? a1 : a0, bfr[j & 3], acc[j]);
  }
#pragma unroll
  for (int j = 0; j < 8; ++j) {
    int rt = wave * 2 + (j >> 2);
    int col = cq * 64 + (j & 3) * 16 + r;
    float g = cent_out[(size_t)m * CH + col];
    float pb = proj_b[col];
#pragma unroll
    for (int v = 0; v < 8; ++v) {
      int row = rt * 16 + v + 8 * hh;
      if (win_mask[(size_t)m * WSZ + row]) {
        float val = acc[j][v] + pb + g;
        atomicAdd(&y[(size_t)win_idx[(size_t)m * WSZ + row] * CH + col], val);
      }
    }
  }
}

__global__ __launch_bounds__(256) void counts_kernel(
    const int* __restrict__ win_idx, const unsigned char* __restrict__ win_mask,
    float* __restrict__ counts, int MW) {
  int i = blockIdx.x * 256 + threadIdx.x;
  if (i < MW && win_mask[i]) atomicAdd(&counts[win_idx[i]], 1.0f);
}

__global__ __launch_bounds__(256) void combine_kernel(
    const float* __restrict__ x, const float* __restrict__ y,
    const float* __restrict__ counts, float* __restrict__ xout, int total) {
  int i = blockIdx.x * 256 + threadIdx.x;
  if (i < total) {
    int row = i / CH;
    xout[i] = x[i] + y[i] / fmaxf(counts[row], 1.f);
  }
}

extern "C" void kernel_launch(void* const* d_in, const int* in_sizes, int n_in,
                              void* d_out, int out_size, void* d_ws, size_t ws_size,
                              hipStream_t stream) {
  const float* x = (const float*)d_in[0];
  const int* win_idx = (const int*)d_in[1];
  const unsigned char* win_mask = (const unsigned char*)d_in[2];
  const float* norm1_g = (const float*)d_in[3];
  const float* norm1_b = (const float*)d_in[4];
  const float* qkv_w = (const float*)d_in[5];
  const float* proj_w = (const float*)d_in[6];
  const float* proj_b = (const float*)d_in[7];
  const float* ca_norm_g = (const float*)d_in[8];
  const float* ca_norm_b = (const float*)d_in[9];
  const float* ca_qkv_w = (const float*)d_in[10];
  const float* ca_qkv_b = (const float*)d_in[11];
  const float* ca_proj_w = (const float*)d_in[12];
  const float* ca_proj_b = (const float*)d_in[13];
  const float* norm2_g = (const float*)d_in[14];
  const float* norm2_b = (const float*)d_in[15];
  const float* ffn_w1 = (const float*)d_in[16];
  const float* ffn_b1 = (const float*)d_in[17];
  const float* ffn_ln_g = (const float*)d_in[18];
  const float* ffn_ln_b = (const float*)d_in[19];
  const float* ffn_w2 = (const float*)d_in[20];
  const float* ffn_b2 = (const float*)d_in[21];

  const int N = in_sizes[0] / CH;
  const int M = in_sizes[1] / WSZ;
  float* out = (float*)d_out;

  char* ws = (char*)d_ws;
  size_t off = 0;
  auto carve = [&](size_t bytes) -> void* {
    void* p = ws + off;
    off = (off + bytes + 255) & ~(size_t)255;
    return p;
  };
  _Float16* xh = (_Float16*)carve((size_t)N * CH * 2);
  _Float16* attn_out = (_Float16*)carve((size_t)M * WSZ * CH * 2);
  float* cent = (float*)carve((size_t)M * CH * 4);
  _Float16* cxh = (_Float16*)carve((size_t)M * CH * 2);
  _Float16* cqkv = (_Float16*)carve((size_t)M * 3 * CH * 2);
  _Float16* co = (_Float16*)carve((size_t)M * CH * 2);
  float* cent_out = (float*)carve((size_t)M * CH * 4);
  float* ybuf = (float*)carve((size_t)N * CH * 4);
  float* counts = (float*)carve((size_t)N * 4);
  _Float16* xh2 = (_Float16*)carve((size_t)N * CH * 2);
  _Float16* hbuf = (_Float16*)carve((size_t)N * HID * 2);
  _Float16* h1 = (_Float16*)carve((size_t)N * HID * 2);
  // packed weights (f16, WMMA B-fragment layout)
  _Float16* qkvw_p   = (_Float16*)carve((size_t)CH * 3 * CH * 2);
  _Float16* projw_p  = (_Float16*)carve((size_t)CH * CH * 2);
  _Float16* caqkvw_p = (_Float16*)carve((size_t)CH * 3 * CH * 2);
  _Float16* caprojw_p= (_Float16*)carve((size_t)CH * CH * 2);
  _Float16* ffnw1_p  = (_Float16*)carve((size_t)CH * HID * 2);
  _Float16* ffnw2_p  = (_Float16*)carve((size_t)HID * CH * 2);

  hipMemsetAsync(ybuf, 0, (size_t)N * CH * 4, stream);
  hipMemsetAsync(counts, 0, (size_t)N * 4, stream);

  // pack all weights into WMMA fragment layout (L2-resident, ~1.6MB total)
  pack_b_kernel<<<(CH / 32) * (3 * CH / 16), 256, 0, stream>>>(qkv_w, qkvw_p, CH, 3 * CH);
  pack_b_kernel<<<(CH / 32) * (CH / 16), 256, 0, stream>>>(proj_w, projw_p, CH, CH);
  pack_b_kernel<<<(CH / 32) * (3 * CH / 16), 256, 0, stream>>>(ca_qkv_w, caqkvw_p, CH, 3 * CH);
  pack_b_kernel<<<(CH / 32) * (CH / 16), 256, 0, stream>>>(ca_proj_w, caprojw_p, CH, CH);
  pack_b_kernel<<<(CH / 32) * (HID / 16), 256, 0, stream>>>(ffn_w1, ffnw1_p, CH, HID);
  pack_b_kernel<<<(HID / 32) * (CH / 16), 256, 0, stream>>>(ffn_w2, ffnw2_p, HID, CH);

  // LN1 -> f16
  ln_f32_kernel<<<N, 256, 0, stream>>>(x, norm1_g, norm1_b, xh, CH, 0);
  // fused window attention per (m, h)
  win_attn_kernel<<<M * NH, 256, 0, stream>>>(xh, win_idx, win_mask, qkvw_p,
                                              attn_out, M);
  // centroid stream
  cent_kernel<<<M, 256, 0, stream>>>(xh, win_idx, win_mask, cent, M);
  ln_f32_kernel<<<M, 256, 0, stream>>>(cent, ca_norm_g, ca_norm_b, cxh, CH, 0);
  gemm_f16out_kernel<<<dim3((M + 63) / 64, 3 * CH / 64), 256, 0, stream>>>(
      cxh, caqkvw_p, ca_qkv_b, cqkv, M, CH, 3 * CH);
  cattn_kernel<<<NH, 256, 0, stream>>>(cqkv, co, M);
  gemm_f32res_kernel<<<dim3((M + 63) / 64, CH / 64), 256, 0, stream>>>(
      co, caprojw_p, ca_proj_b, cent, cent_out, M, CH, CH);
  // window proj + global ctx + scatter-add
  win_proj_scatter_kernel<<<dim3(M, CH / 64), 256, 0, stream>>>(
      attn_out, projw_p, proj_b, cent_out, win_idx, win_mask, ybuf, M);
  counts_kernel<<<(M * WSZ + 255) / 256, 256, 0, stream>>>(win_idx, win_mask,
                                                           counts, M * WSZ);
  // x_out = x + y / max(counts, 1)
  combine_kernel<<<((size_t)N * CH + 255) / 256, 256, 0, stream>>>(
      x, ybuf, counts, out, N * CH);
  // FFN
  ln_f32_kernel<<<N, 256, 0, stream>>>(out, norm2_g, norm2_b, xh2, CH, 0);
  gemm_f16out_kernel<<<dim3((N + 63) / 64, HID / 64), 256, 0, stream>>>(
      xh2, ffnw1_p, ffn_b1, hbuf, N, CH, HID);
  ln_f16_kernel<<<N, 256, 0, stream>>>(hbuf, ffn_ln_g, ffn_ln_b, h1, HID, 1);
  gemm_f32res_kernel<<<dim3((N + 63) / 64, CH / 64), 256, 0, stream>>>(
      h1, ffnw2_p, ffn_b2, out, out, N, HID, CH);
}
